// Inter_attention_21131239096480
// MI455X (gfx1250) — matrix operand order repaired and checked by reading, not compiled, not run on hardware
//
#include <hip/hip_runtime.h>

#define N_NODES 20000
#define N_EDGES 320000
#define CZ 256
#define CE 64
#define DF 576          // 2*CZ + CE
#define DF_H 288        // half row for LN split
#define HC 256          // HEADS * C_OUT
#define DH 512          // MLP hidden
#define LN_EPS 1e-5f
#define INV_SQRT_CO 0.17677669529663689f

typedef __attribute__((ext_vector_type(16))) __bf16 v16bf;
typedef __attribute__((ext_vector_type(8)))  __bf16 v8bf;
typedef __attribute__((ext_vector_type(8)))  float  v8f;

union BF16x16 { v16bf v; v8bf h[2]; };

__device__ __forceinline__ v8f wmma_bf16(v16bf a, v16bf b, v8f c) {
    // D = A(16x32 bf16) * B(32x16 bf16) + C(16x16 f32)
    return __builtin_amdgcn_wmma_f32_16x16x32_bf16(false, a, false, b, (short)0, c, false, false);
}

// A fragment: LDS row-major bf16, row stride = strideK elems (16B-aligned).
__device__ __forceinline__ v16bf load_afrag(const __bf16* lds, int strideK, int kk, int l, int hi) {
    const __bf16* p = lds + l * strideK + kk + hi * 8;
    BF16x16 t;
    t.h[0] = *(const v8bf*)(p);
    t.h[1] = *(const v8bf*)(p + 16);
    return t.v;
}

// B fragment: pre-packed so each lane's 16 bf16 are contiguous (32B).
__device__ __forceinline__ v16bf load_bfrag(const __bf16* pk) {
    BF16x16 t;
    t.h[0] = *(const v8bf*)(pk);
    t.h[1] = *(const v8bf*)(pk + 8);
    return t.v;
}

// LDS byte offset of a shared-memory derived pointer (low 32 bits of generic addr)
__device__ __forceinline__ unsigned lds_off(const void* p) {
    return (unsigned)(uintptr_t)p;
}

// CDNA5 async global->LDS copy, 16 bytes per lane, ASYNCcnt-tracked
__device__ __forceinline__ void async_load_b128(unsigned ldsaddr, const void* gaddr) {
    asm volatile("global_load_async_to_lds_b128 %0, %1, off"
                 :: "v"(ldsaddr), "v"(gaddr) : "memory");
}
// CDNA5 async LDS->global copy, 16 bytes per lane
__device__ __forceinline__ void async_store_b128(const void* gaddr, unsigned ldsaddr) {
    asm volatile("global_store_async_from_lds_b128 %0, %1, off"
                 :: "v"(gaddr), "v"(ldsaddr) : "memory");
}
__device__ __forceinline__ void wait_async0() {
    asm volatile("s_wait_asynccnt 0x0" ::: "memory");
}

// ---- monotone float<->uint encoding for atomic segment-max ----
__device__ __forceinline__ unsigned fenc(float f) {
    unsigned u = __float_as_uint(f);
    return (u & 0x80000000u) ? ~u : (u | 0x80000000u);
}
__device__ __forceinline__ float fdec(unsigned u) {
    unsigned b = (u & 0x80000000u) ? (u & 0x7FFFFFFFu) : ~u;
    return __uint_as_float(b);
}

// ================= K0: pack a KxN f32 weight into WMMA-B fragment order =================
// B layout (bf16 32x16): lanes 0-15 hold K=0..7,16..23 (N=lane), lanes 16-31 hold K=8..15,24..31.
__global__ void pack_weight_kernel(const float* __restrict__ W, __bf16* __restrict__ dst,
                                   int K, int Nn) {
    int i = blockIdx.x * 256 + threadIdx.x;
    if (i >= K * Nn) return;
    int k = i / Nn, n = i - k * Nn;
    int kb = k >> 5, kr = k & 31;
    int nb = n >> 4, nl = n & 15;
    int hi = (kr >> 3) & 1;
    int e  = (kr & 7) | ((kr >> 4) << 3);
    int lane = nl | (hi << 4);
    int NB = Nn >> 4;
    dst[(((size_t)(kb * NB + nb) * 32) + lane) * 16 + e] = (__bf16)W[i];
}

// ================= K1: q = x @ Wq + bq  (WMMA) =================
__global__ __launch_bounds__(32) void proj_q_kernel(const float* __restrict__ x,
                                                    const __bf16* __restrict__ wqP,
                                                    const float* __restrict__ bq,
                                                    float* __restrict__ qbuf) {
    __shared__ __align__(16) __bf16 aB[16 * CZ];
    int tid = threadIdx.x;
    int m0 = blockIdx.x * 16;
    for (int i = tid; i < 16 * CZ; i += 32)
        aB[i] = (__bf16)x[(size_t)m0 * CZ + i];
    __syncthreads();

    int l = tid & 15, hi = tid >> 4;
    v8f acc[16] = {};
    for (int kk = 0; kk < CZ; kk += 32) {
        v16bf af = load_afrag(aB, CZ, kk, l, hi);
        int kb = kk >> 5;
        if (kk + 32 < CZ) __builtin_prefetch(wqP + ((size_t)(kb + 1) * 16 * 32) * 16, 0, 1);
#pragma unroll
        for (int nb = 0; nb < 16; ++nb) {
            v16bf bf = load_bfrag(wqP + (((size_t)(kb * 16 + nb) * 32) + tid) * 16);
            acc[nb] = wmma_bf16(af, bf, acc[nb]);
        }
    }
#pragma unroll
    for (int nb = 0; nb < 16; ++nb) {
        int n = nb * 16 + l;
        float b = bq[n];
#pragma unroll
        for (int r = 0; r < 8; ++r)
            qbuf[(size_t)(m0 + r + hi * 8) * HC + n] = acc[nb][r] + b;
    }
}

// ================= K2: fused edge kernel =================
// async gather -> LN -> k-GEMM + scores -> eg-GEMM -> v-GEMM -> async store of v*eg
__global__ __launch_bounds__(32) void edge_kv_kernel(
    const float* __restrict__ x, const float* __restrict__ p,
    const float* __restrict__ ea, const int* __restrict__ ei,
    const float* __restrict__ lnw, const float* __restrict__ lnb,
    const float* __restrict__ bk, const float* __restrict__ bv,
    const __bf16* __restrict__ wkP, const __bf16* __restrict__ wvP,
    const __bf16* __restrict__ weP,
    const float* __restrict__ qbuf,
    float* __restrict__ scoresbuf, __bf16* __restrict__ wbuf) {

    __shared__ __align__(16) char smem[57728];
    float*  featF = (float*)smem;                  // [16][576] f32; later reused as ldsT [16][256] f32
    __bf16* featB = (__bf16*)(smem + 36864);       // [16][576] bf16 (post-LN); later w-staging [16][256] bf16
    __bf16* eaB   = (__bf16*)(smem + 55296);       // [16][64]  bf16 raw edge_attr
    int*    idxS  = (int*)(smem + 57344);          // [32] src(0..15), dst(16..31)
    float*  pSum  = (float*)(smem + 57472);        // [32] LN partial sums
    float*  pSum2 = (float*)(smem + 57600);        // [32] LN partial sums of squares

    int tid = threadIdx.x;
    int e0 = blockIdx.x * 16;

    if (tid < 16) {
        idxS[tid]      = ei[e0 + tid];             // src
        idxS[16 + tid] = ei[N_EDGES + e0 + tid];   // dst
    }
    __syncthreads();

    // ---- async gather of concat(edge_attr, x[src], p[dst]) straight into LDS ----
    // 16 bytes (4 f32) per lane per iteration; segment boundaries (64, 320) are x4 aligned.
    for (int i4 = tid; i4 < (16 * DF) / 4; i4 += 32) {
        int i = i4 * 4;
        int r = i / DF, c = i - r * DF;
        const float* gp = (c < CE)
            ? (ea + (size_t)(e0 + r) * CE + c)
            : ((c < CE + CZ) ? (x + (size_t)idxS[r] * CZ + (c - CE))
                             : (p + (size_t)idxS[16 + r] * CZ + (c - CE - CZ)));
        async_load_b128(lds_off(featF + i), gp);
    }
    wait_async0();
    __syncthreads();

    int l = tid & 15, hi = tid >> 4;

    // ---- copy raw edge_attr tile to bf16 (A operand of eg-GEMM) ----
    for (int i = tid; i < 16 * CE; i += 32) {
        int r = i >> 6, c = i & 63;
        eaB[i] = (__bf16)featF[r * DF + c];
    }

    // ---- LayerNorm: lane pair (l, l+16) shares row l, each lane does 288 cols ----
    {
        const float* row = featF + l * DF;
        float s = 0.f, s2 = 0.f;
        for (int c = hi * DF_H; c < hi * DF_H + DF_H; ++c) {
            float v = row[c]; s += v; s2 += v * v;
        }
        pSum[tid] = s; pSum2[tid] = s2;
    }
    __syncthreads();
    {
        float s  = pSum[l] + pSum[l + 16];
        float s2 = pSum2[l] + pSum2[l + 16];
        float mu  = s * (1.0f / DF);
        float var = s2 * (1.0f / DF) - mu * mu;
        float inv = rsqrtf(var + LN_EPS);
        const float* row = featF + l * DF;
        __bf16* outp = featB + l * DF;
        for (int c = hi * DF_H; c < hi * DF_H + DF_H; ++c)
            outp[c] = (__bf16)((row[c] - mu) * inv * lnw[c] + lnb[c]);
    }
    __syncthreads();

    float* ldsT = featF;   // featF region now free -> k / eg staging tile [16][256] f32

    // ---- k = LN(feat) @ Wk + bk ----
    {
        v8f acc[16] = {};
        for (int kk = 0; kk < DF; kk += 32) {
            v16bf af = load_afrag(featB, DF, kk, l, hi);
            int kb = kk >> 5;
            if (kk + 32 < DF) __builtin_prefetch(wkP + ((size_t)(kb + 1) * 16 * 32) * 16, 0, 1);
#pragma unroll
            for (int nb = 0; nb < 16; ++nb) {
                v16bf bf = load_bfrag(wkP + (((size_t)(kb * 16 + nb) * 32) + tid) * 16);
                acc[nb] = wmma_bf16(af, bf, acc[nb]);
            }
        }
#pragma unroll
        for (int nb = 0; nb < 16; ++nb) {
            int n = nb * 16 + l;
            float b = bk[n];
#pragma unroll
            for (int r = 0; r < 8; ++r)
                ldsT[(r + hi * 8) * HC + n] = acc[nb][r] + b;
        }
    }
    __syncthreads();

    // ---- scores[m,h] = <q[src[m],h,:], k[m,h,:]> / sqrt(32) ----
    for (int j = 0; j < 4; ++j) {
        int pidx = tid + 32 * j;          // 0..127 -> (m,h)
        int m = pidx >> 3, h = pidx & 7;
        const float* qrow = qbuf + (size_t)idxS[m] * HC + h * 32;
        const float* krow = ldsT + m * HC + h * 32;
        float s = 0.f;
        for (int c = 0; c < 32; ++c) s += qrow[c] * krow[c];
        scoresbuf[(size_t)(e0 + m) * 8 + h] = s * INV_SQRT_CO;
    }
    __syncthreads();

    // ---- eg = edge_attr @ We  (stage into ldsT) ----
    {
        v8f acc[16] = {};
        for (int kk = 0; kk < CE; kk += 32) {
            v16bf af = load_afrag(eaB, CE, kk, l, hi);
            int kb = kk >> 5;
#pragma unroll
            for (int nb = 0; nb < 16; ++nb) {
                v16bf bf = load_bfrag(weP + (((size_t)(kb * 16 + nb) * 32) + tid) * 16);
                acc[nb] = wmma_bf16(af, bf, acc[nb]);
            }
        }
#pragma unroll
        for (int nb = 0; nb < 16; ++nb) {
            int n = nb * 16 + l;
#pragma unroll
            for (int r = 0; r < 8; ++r)
                ldsT[(r + hi * 8) * HC + n] = acc[nb][r];
        }
    }
    __syncthreads();

    // ---- v = LN(feat) @ Wv + bv ;  w = v * eg -> LDS stage -> async coalesced store ----
    {
        v8f acc[16] = {};
        for (int kk = 0; kk < DF; kk += 32) {
            v16bf af = load_afrag(featB, DF, kk, l, hi);
            int kb = kk >> 5;
            if (kk + 32 < DF) __builtin_prefetch(wvP + ((size_t)(kb + 1) * 16 * 32) * 16, 0, 1);
#pragma unroll
            for (int nb = 0; nb < 16; ++nb) {
                v16bf bf = load_bfrag(wvP + (((size_t)(kb * 16 + nb) * 32) + tid) * 16);
                acc[nb] = wmma_bf16(af, bf, acc[nb]);
            }
        }
        __bf16* wB = featB;   // featB dead after last A-fragment read above
#pragma unroll
        for (int nb = 0; nb < 16; ++nb) {
            int n = nb * 16 + l;
            float b = bv[n];
#pragma unroll
            for (int r = 0; r < 8; ++r) {
                int m = r + hi * 8;
                float w = (acc[nb][r] + b) * ldsT[m * HC + n];
                wB[m * HC + n] = (__bf16)w;
            }
        }
        __syncthreads();
        // drain 16x256 bf16 tile (8 KB) with 512x 16B async chunks, fully coalesced
        for (int j = 0; j < 16; ++j) {
            int ch = tid + 32 * j;     // 0..511
            int m  = ch >> 5;          // 32 chunks of 16B per 512B row
            int cb = ch & 31;
            const __bf16* gp = wbuf + (size_t)(e0 + m) * HC + cb * 8;
            async_store_b128(gp, lds_off(wB + m * HC + cb * 8));
        }
        wait_async0();
    }
}

// ================= K3: init segment buffers =================
__global__ void seg_init_kernel(unsigned* __restrict__ smax, float* __restrict__ ssum,
                                float* __restrict__ accb) {
    int i = blockIdx.x * 256 + threadIdx.x;
    if (i < N_NODES * 8) { smax[i] = 0u; ssum[i] = 0.f; }
    if (i < N_NODES * HC) accb[i] = 0.f;
}

// ================= K4: segment max over src =================
__global__ void seg_max_kernel(const float* __restrict__ scores, const int* __restrict__ ei,
                               unsigned* __restrict__ smax) {
    int i = blockIdx.x * 256 + threadIdx.x;
    if (i >= N_EDGES * 8) return;
    int e = i >> 3, h = i & 7;
    int node = ei[e];
    atomicMax(&smax[node * 8 + h], fenc(scores[i]));
}

// ================= K5: exp + segment sum =================
__global__ void seg_exp_kernel(float* __restrict__ scores, const int* __restrict__ ei,
                               const unsigned* __restrict__ smax, float* __restrict__ ssum) {
    int i = blockIdx.x * 256 + threadIdx.x;
    if (i >= N_EDGES * 8) return;
    int e = i >> 3, h = i & 7;
    int node = ei[e];
    float m = fdec(smax[node * 8 + h]);
    float ex = __expf(scores[i] - m);
    scores[i] = ex;
    atomicAdd(&ssum[node * 8 + h], ex);
}

// ================= K6: weighted scatter-sum =================
__global__ void aggregate_kernel(const float* __restrict__ exb, const float* __restrict__ ssum,
                                 const int* __restrict__ ei, const __bf16* __restrict__ wbuf,
                                 float* __restrict__ accb) {
    long long i = (long long)blockIdx.x * 256 + threadIdx.x;   // over E * HC
    int e = (int)(i >> 8);
    int n = (int)(i & 255);
    int h = n >> 5;
    int node = ei[e];
    float alpha = exb[(size_t)e * 8 + h] / ssum[node * 8 + h];
    atomicAdd(&accb[(size_t)node * HC + n], alpha * (float)wbuf[i]);
}

// ================= K7: fused MLP: silu(acc@W1+b1)@W2 + b2 =================
__global__ __launch_bounds__(32) void mlp_kernel(const float* __restrict__ accb,
                                                 const __bf16* __restrict__ w1P,
                                                 const float* __restrict__ b1,
                                                 const __bf16* __restrict__ w2P,
                                                 const float* __restrict__ b2,
                                                 float* __restrict__ out) {
    __shared__ __align__(16) __bf16 aB[16 * CZ];   // 8 KB
    __shared__ __align__(16) __bf16 hB[16 * DH];   // 16 KB
    int tid = threadIdx.x;
    int m0 = blockIdx.x * 16;
    for (int i = tid; i < 16 * CZ; i += 32)
        aB[i] = (__bf16)accb[(size_t)m0 * CZ + i];
    __syncthreads();

    int l = tid & 15, hi = tid >> 4;

    // GEMM1 (N=512) in two 256-wide halves, SiLU, stage bf16 to LDS
    for (int half = 0; half < 2; ++half) {
        v8f acc[16] = {};
        for (int kk = 0; kk < CZ; kk += 32) {
            v16bf af = load_afrag(aB, CZ, kk, l, hi);
            int kb = kk >> 5;
#pragma unroll
            for (int nb = 0; nb < 16; ++nb) {
                v16bf bf = load_bfrag(w1P + (((size_t)(kb * 32 + half * 16 + nb) * 32) + tid) * 16);
                acc[nb] = wmma_bf16(af, bf, acc[nb]);
            }
        }
#pragma unroll
        for (int nb = 0; nb < 16; ++nb) {
            int n = (half * 16 + nb) * 16 + l;
            float b = b1[n];
#pragma unroll
            for (int r = 0; r < 8; ++r) {
                float hv = acc[nb][r] + b;
                float sv = hv / (1.f + __expf(-hv));   // SiLU
                hB[(r + hi * 8) * DH + n] = (__bf16)sv;
            }
        }
    }
    __syncthreads();

    // GEMM2: K=512 -> N=256
    v8f acc2[16] = {};
    for (int kk = 0; kk < DH; kk += 32) {
        v16bf af = load_afrag(hB, DH, kk, l, hi);
        int kb = kk >> 5;
        if (kk + 32 < DH) __builtin_prefetch(w2P + ((size_t)(kb + 1) * 16 * 32) * 16, 0, 1);
#pragma unroll
        for (int nb = 0; nb < 16; ++nb) {
            v16bf bf = load_bfrag(w2P + (((size_t)(kb * 16 + nb) * 32) + tid) * 16);
            acc2[nb] = wmma_bf16(af, bf, acc2[nb]);
        }
    }
#pragma unroll
    for (int nb = 0; nb < 16; ++nb) {
        int n = nb * 16 + l;
        float b = b2[n];
#pragma unroll
        for (int r = 0; r < 8; ++r)
            out[(size_t)(m0 + r + hi * 8) * CZ + n] = acc2[nb][r] + b;
    }
}

// ================= workspace layout (bytes) =================
static const size_t OFF_Q    = 0;            // q       : N*256 f32   = 20,480,000
static const size_t OFF_SC   = 20480000;     // scores  : E*8  f32    = 10,240,000
static const size_t OFF_SMX  = 30720000;     // smax    : N*8  u32    =    640,000
static const size_t OFF_SSUM = 31360000;     // ssum    : N*8  f32    =    640,000
static const size_t OFF_ACC  = 32000000;     // acc     : N*256 f32   = 20,480,000
static const size_t OFF_W    = 52480000;     // w=v*eg  : E*256 bf16  = 163,840,000
static const size_t OFF_WQP  = 216320000;    // packed Wq  (131,072)
static const size_t OFF_WKP  = 216451072;    // packed Wk  (294,912)
static const size_t OFF_WVP  = 216745984;    // packed Wv  (294,912)
static const size_t OFF_WEP  = 217040896;    // packed We  (32,768)
static const size_t OFF_W1P  = 217073664;    // packed W1  (262,144)
static const size_t OFF_W2P  = 217335808;    // packed W2  (262,144) -> total 217,597,952 B

extern "C" void kernel_launch(void* const* d_in, const int* in_sizes, int n_in,
                              void* d_out, int out_size, void* d_ws, size_t ws_size,
                              hipStream_t stream) {
    (void)in_sizes; (void)n_in; (void)out_size; (void)ws_size;

    const float* x   = (const float*)d_in[0];
    const float* p   = (const float*)d_in[1];
    const float* ea  = (const float*)d_in[2];
    const int*   ei  = (const int*)d_in[3];
    const float* lnw = (const float*)d_in[4];
    const float* lnb = (const float*)d_in[5];
    const float* Wq  = (const float*)d_in[6];
    const float* bq  = (const float*)d_in[7];
    const float* Wk  = (const float*)d_in[8];
    const float* bk  = (const float*)d_in[9];
    const float* Wv  = (const float*)d_in[10];
    const float* bv  = (const float*)d_in[11];
    const float* We  = (const float*)d_in[12];
    const float* W1  = (const float*)d_in[13];
    const float* b1  = (const float*)d_in[14];
    const float* W2  = (const float*)d_in[15];
    const float* b2  = (const float*)d_in[16];
    float* out = (float*)d_out;

    char* ws = (char*)d_ws;
    float*    qbuf   = (float*)(ws + OFF_Q);
    float*    scores = (float*)(ws + OFF_SC);
    unsigned* smax   = (unsigned*)(ws + OFF_SMX);
    float*    ssum   = (float*)(ws + OFF_SSUM);
    float*    accb   = (float*)(ws + OFF_ACC);
    __bf16*   wbuf   = (__bf16*)(ws + OFF_W);
    __bf16*   wqP    = (__bf16*)(ws + OFF_WQP);
    __bf16*   wkP    = (__bf16*)(ws + OFF_WKP);
    __bf16*   wvP    = (__bf16*)(ws + OFF_WVP);
    __bf16*   weP    = (__bf16*)(ws + OFF_WEP);
    __bf16*   w1P    = (__bf16*)(ws + OFF_W1P);
    __bf16*   w2P    = (__bf16*)(ws + OFF_W2P);

    // K0: pack all weights to bf16 WMMA-B fragment layout (lives in L2 afterwards)
    pack_weight_kernel<<<dim3((CZ * HC + 255) / 256), 256, 0, stream>>>(Wq, wqP, CZ, HC);
    pack_weight_kernel<<<dim3((DF * HC + 255) / 256), 256, 0, stream>>>(Wk, wkP, DF, HC);
    pack_weight_kernel<<<dim3((DF * HC + 255) / 256), 256, 0, stream>>>(Wv, wvP, DF, HC);
    pack_weight_kernel<<<dim3((CE * HC + 255) / 256), 256, 0, stream>>>(We, weP, CE, HC);
    pack_weight_kernel<<<dim3((CZ * DH + 255) / 256), 256, 0, stream>>>(W1, w1P, CZ, DH);
    pack_weight_kernel<<<dim3((DH * CZ + 255) / 256), 256, 0, stream>>>(W2, w2P, DH, CZ);

    // K1: q projection (WMMA)
    proj_q_kernel<<<dim3(N_NODES / 16), 32, 0, stream>>>(x, wqP, bq, qbuf);

    // K3: init segment buffers
    seg_init_kernel<<<dim3((N_NODES * HC) / 256), 256, 0, stream>>>(smax, ssum, accb);

    // K2: fused edge gather+LN+k/v/eg GEMMs + raw scores (WMMA + async LDS paths)
    edge_kv_kernel<<<dim3(N_EDGES / 16), 32, 0, stream>>>(
        x, p, ea, ei, lnw, lnb, bk, bv, wkP, wvP, weP, qbuf, scores, wbuf);

    // K4/K5: scatter softmax over src
    seg_max_kernel<<<dim3((N_EDGES * 8) / 256), 256, 0, stream>>>(scores, ei, smax);
    seg_exp_kernel<<<dim3((N_EDGES * 8) / 256), 256, 0, stream>>>(scores, ei, smax, ssum);

    // K6: alpha-weighted scatter-sum of v*eg
    aggregate_kernel<<<dim3((N_EDGES * HC) / 256), 256, 0, stream>>>(scores, ssum, ei, wbuf, accb);

    // K7: fused output MLP (WMMA)
    mlp_kernel<<<dim3(N_NODES / 16), 32, 0, stream>>>(accb, w1P, b1, w2P, b2, out);
}
